// Attention_21715354649706
// MI455X (gfx1250) — matrix operand order repaired and checked by reading, not compiled, not run on hardware
//
#include <hip/hip_runtime.h>
#include <hip/hip_bf16.h>
#include <stdint.h>

#define DIM    768
#define NHEADS 12
#define HDIM   64
#define SEQ    2048
#define BATCH  4
#define BN     (BATCH * SEQ)
#define QKV_O  (3 * DIM)
#define SCALE  0.125f   // 64^-0.5

typedef __bf16 bf16x16 __attribute__((ext_vector_type(16)));
typedef float  f32x8   __attribute__((ext_vector_type(8)));

union Frag {
    bf16x16 v;
    uint4 q[2];
    unsigned short s[16];
};

__device__ __forceinline__ unsigned short f2bf(float f) {
    unsigned int u = __float_as_uint(f);
    u += 0x7FFFu + ((u >> 16) & 1u);   // round to nearest even
    return (unsigned short)(u >> 16);
}

__device__ __forceinline__ f32x8 wmma_bf16(const Frag& a, const Frag& b, f32x8 c) {
    return __builtin_amdgcn_wmma_f32_16x16x32_bf16(false, a.v, false, b.v,
                                                   (short)0, c, false, false);
}

// CDNA5 async copy: 16 bytes global -> LDS, tracked by ASYNCcnt
__device__ __forceinline__ void async_copy_b128(const unsigned short* gsrc,
                                                const unsigned short* ldst) {
    unsigned loff = (unsigned)(uintptr_t)ldst;   // low 32 bits = LDS address
    asm volatile("global_load_async_to_lds_b128 %0, %1, off"
                 :: "v"(loff), "v"((unsigned long long)(uintptr_t)gsrc)
                 : "memory");
}

__device__ __forceinline__ void wait_asynccnt0() {
    asm volatile("s_wait_asynccnt 0x0" ::: "memory");
}

// Load two 16-byte chunks (one 32x16 B-fragment lane's share) at p, p+16
__device__ __forceinline__ Frag load_frag32B(const unsigned short* p) {
    Frag f;
    f.q[0] = *reinterpret_cast<const uint4*>(p);
    f.q[1] = *reinterpret_cast<const uint4*>(p + 16);
    return f;
}

// A fragment: 16x32 bf16 from row-major src (leading dim ld) at (row0, k0).
__device__ __forceinline__ Frag load_a_global(const unsigned short* src, int ld,
                                              int row0, int k0, int lane) {
    Frag f;
    int m = lane & 15, half = lane >> 4;
    const unsigned short* p = src + (size_t)(row0 + m) * ld + k0 + half * 8;
    f.q[0] = *reinterpret_cast<const uint4*>(p);
    f.q[1] = *reinterpret_cast<const uint4*>(p + 16);
    return f;
}

// ---------------------------------------------------------------- cast kernel
__global__ void cast_f32_bf16(const float* __restrict__ in,
                              unsigned short* __restrict__ out, int n) {
    int i = blockIdx.x * blockDim.x + threadIdx.x;
    int stride = gridDim.x * blockDim.x;
    for (; i < n; i += stride) out[i] = f2bf(in[i]);
}

// ============================================================================
// GEMM core: 16-row tile per block, 64 cols per wave (4 waves / block).
// A tile staged through LDS with async-to-LDS, two K-steps (a "pair") at a
// time so all 128 threads issue exactly one async copy (no divergence).
// Two pair-buffers ping-pong; copies for pair i+1 overlap WMMAs of pair i.
// B fragments load from one hoisted per-lane base pointer + immediate offsets.
// ============================================================================
__device__ __forceinline__ void gemm_stage_pair(const unsigned short* srcA,
                                                unsigned short* lds,
                                                int row0, int k0, int pair) {
    int t = threadIdx.x;
    int sub = t >> 6, tt = t & 63;          // sub: which K-step of the pair
    int rr = tt >> 2, pp = tt & 3;          // row, 16B chunk within row
    async_copy_b128(srcA + (size_t)(row0 + rr) * DIM + k0 + sub * 32 + pp * 8,
                    lds + (((pair * 2 + sub) * 16) + rr) * 32 + pp * 8);
}

__device__ __forceinline__ void gemm_tile(const unsigned short* __restrict__ srcA,
                                          const unsigned short* __restrict__ srcW,
                                          unsigned short* lds,
                                          int row0, int col0, int lane,
                                          f32x8 acc[4]) {
    int n = lane & 15, half = lane >> 4;
    // invariant per-lane bases; all per-step deltas become immediate offsets
    const unsigned short* wbase = srcW + (size_t)(col0 + n) * DIM + half * 8;
    const unsigned short* abase = lds + (size_t)(lane & 15) * 32 + half * 8;

    gemm_stage_pair(srcA, lds, row0, 0, 0);
    #pragma unroll
    for (int k0 = 0; k0 < DIM; k0 += 64) {
        int pair = (k0 >> 6) & 1;
        wait_asynccnt0();
        __syncthreads();
        if (k0 + 64 < DIM) gemm_stage_pair(srcA, lds, row0, k0 + 64, pair ^ 1);
        #pragma unroll
        for (int sub = 0; sub < 2; ++sub) {
            Frag a;
            const unsigned short* ap = abase + (pair * 2 + sub) * 16 * 32;
            a.q[0] = *reinterpret_cast<const uint4*>(ap);
            a.q[1] = *reinterpret_cast<const uint4*>(ap + 16);
            #pragma unroll
            for (int t = 0; t < 4; ++t) {
                Frag b = load_frag32B(wbase + (size_t)t * 16 * DIM + k0 + sub * 32);
                acc[t] = wmma_bf16(a, b, acc[t]);
            }
        }
    }
}

// ------------------------------------------------------------- QKV GEMM kernel
// out = x @ w_qkv^T, scattered into per-head Q(scaled)/K/V [B,H,N,D] bf16
__global__ void qkv_gemm(const unsigned short* __restrict__ xb,
                         const unsigned short* __restrict__ wb,
                         unsigned short* __restrict__ Qb,
                         unsigned short* __restrict__ Kb,
                         unsigned short* __restrict__ Vb) {
    __shared__ __align__(16) unsigned short ldsA[2 * 2 * 16 * 32];  // 4 KB
    int wave = threadIdx.x >> 5, lane = threadIdx.x & 31;
    int row0 = blockIdx.x * 16;
    int colg = __builtin_amdgcn_readfirstlane(blockIdx.y * 4 + wave); // [0,36)
    int col0 = colg * 64;

    f32x8 acc[4] = {};
    gemm_tile(xb, wb, ldsA, row0, col0, lane, acc);

    int three = colg / NHEADS, h = colg % NHEADS;
    unsigned short* dst = (three == 0) ? Qb : (three == 1) ? Kb : Vb;
    float scl = (three == 0) ? SCALE : 1.0f;
    int n = lane & 15, half = lane >> 4;
    #pragma unroll
    for (int t = 0; t < 4; ++t) {
        #pragma unroll
        for (int r = 0; r < 8; ++r) {
            int row = row0 + r + 8 * half;
            int bi = row >> 11, nn = row & (SEQ - 1);   // SEQ = 2048
            int d = t * 16 + n;
            dst[(((size_t)bi * NHEADS + h) * SEQ + nn) * HDIM + d] =
                f2bf(acc[t][r] * scl);
        }
    }
}

// --------------------------------------------------------- flash attention
// One wave per 16-query tile; 4 waves/block share LDS-transposed V tiles.
__global__ void flash_attn(const unsigned short* __restrict__ Qb,
                           const unsigned short* __restrict__ Kb,
                           const unsigned short* __restrict__ Vb,
                           unsigned short* __restrict__ Ob) {
    __shared__ __align__(16) unsigned short ldsVt[HDIM][40]; // V^T, padded rows
    __shared__ __align__(16) unsigned short ldsP[4][16][40]; // per-wave P tile

    int wave = threadIdx.x >> 5, lane = threadIdx.x & 31;
    int h = blockIdx.y, b = blockIdx.z;
    size_t bh = (size_t)b * NHEADS + h;
    int q0 = (blockIdx.x * 4 + wave) * 16;
    const unsigned short* Qp = Qb + bh * SEQ * HDIM;
    const unsigned short* Kp = Kb + bh * SEQ * HDIM;
    const unsigned short* Vp = Vb + bh * SEQ * HDIM;

    // Q fragments (d-dim split 0..31 / 32..63), reused over all key tiles
    Frag aq0 = load_a_global(Qp, HDIM, q0, 0, lane);
    Frag aq1 = load_a_global(Qp, HDIM, q0, 32, lane);

    f32x8 Oacc[4] = {};
    float m[8], l[8];
    #pragma unroll
    for (int r = 0; r < 8; ++r) { m[r] = -1e30f; l[r] = 0.f; }

    int n = lane & 15, half = lane >> 4;

    for (int key0 = 0; key0 < SEQ; key0 += 32) {
        // prefetch next K/V tiles into cache while this tile computes
        if (key0 + 32 < SEQ) {
            const unsigned short* pk = Kp + (size_t)(key0 + 32 + lane) * HDIM;
            const unsigned short* pv = Vp + (size_t)(key0 + 32 + lane) * HDIM;
            __builtin_prefetch(pk, 0, 3);
            __builtin_prefetch(pv, 0, 3);
        }

        __syncthreads();  // previous iteration done reading ldsVt
        // cooperative stage: ldsVt[d][kk] = V[key0+kk][d]
        {
            int kk = threadIdx.x >> 2;           // 0..31
            int c0 = (threadIdx.x & 3) * 16;     // 0,16,32,48
            const unsigned short* vp = Vp + (size_t)(key0 + kk) * HDIM + c0;
            unsigned short tmp[16];
            *reinterpret_cast<uint4*>(tmp)     = *reinterpret_cast<const uint4*>(vp);
            *reinterpret_cast<uint4*>(tmp + 8) = *reinterpret_cast<const uint4*>(vp + 8);
            #pragma unroll
            for (int i = 0; i < 16; ++i) ldsVt[c0 + i][kk] = tmp[i];
        }
        __syncthreads();

        // scores S[16 x 32] = Q_tile . K_tile^T  (4 WMMAs)
        // one hoisted base; fragment deltas (jc*16 rows, k0) are immediates
        f32x8 s0 = {}, s1 = {};
        {
            const unsigned short* kb =
                Kp + (size_t)(key0 + n) * HDIM + half * 8;
            Frag bk00 = load_frag32B(kb);
            Frag bk01 = load_frag32B(kb + 32);
            Frag bk10 = load_frag32B(kb + 16 * HDIM);
            Frag bk11 = load_frag32B(kb + 16 * HDIM + 32);
            s0 = wmma_bf16(aq0, bk00, s0);
            s0 = wmma_bf16(aq1, bk01, s0);
            s1 = wmma_bf16(aq0, bk10, s1);
            s1 = wmma_bf16(aq1, bk11, s1);
        }

        // online softmax (row = r + 8*half; cols spread over 16-lane half)
        #pragma unroll
        for (int r = 0; r < 8; ++r) {
            float v0 = s0[r], v1 = s1[r];
            float mx = fmaxf(v0, v1);
            mx = fmaxf(mx, __shfl_xor(mx, 1, 32));
            mx = fmaxf(mx, __shfl_xor(mx, 2, 32));
            mx = fmaxf(mx, __shfl_xor(mx, 4, 32));
            mx = fmaxf(mx, __shfl_xor(mx, 8, 32));
            float mnew  = fmaxf(m[r], mx);
            float alpha = __expf(m[r] - mnew);
            float p0 = __expf(v0 - mnew);
            float p1 = __expf(v1 - mnew);
            float rs = p0 + p1;
            rs += __shfl_xor(rs, 1, 32);
            rs += __shfl_xor(rs, 2, 32);
            rs += __shfl_xor(rs, 4, 32);
            rs += __shfl_xor(rs, 8, 32);
            l[r] = l[r] * alpha + rs;
            m[r] = mnew;
            #pragma unroll
            for (int c = 0; c < 4; ++c) Oacc[c][r] *= alpha;
            int mrow = r + 8 * half;
            ldsP[wave][mrow][n]      = f2bf(p0);
            ldsP[wave][mrow][16 + n] = f2bf(p1);
        }
        // wave-private LDS region: wait on DS counter before re-reading
        asm volatile("s_wait_dscnt 0x0" ::: "memory");

        // P as A-fragment (16x32 bf16)
        Frag pf;
        {
            const unsigned short* pp = &ldsP[wave][lane & 15][half * 8];
            pf.q[0] = *reinterpret_cast<const uint4*>(pp);
            pf.q[1] = *reinterpret_cast<const uint4*>(pp + 16);
        }
        // O[16x64] += P[16x32] . V[32x64]  (4 WMMAs from transposed LDS)
        #pragma unroll
        for (int c = 0; c < 4; ++c) {
            Frag bv;
            const unsigned short* vp = &ldsVt[c * 16 + n][half * 8];
            bv.q[0] = *reinterpret_cast<const uint4*>(vp);
            bv.q[1] = *reinterpret_cast<const uint4*>(vp + 16);
            Oacc[c] = wmma_bf16(pf, bv, Oacc[c]);
        }
    }

    // normalize and store bf16 into [B,N,C] layout for the proj GEMM
    #pragma unroll
    for (int r = 0; r < 8; ++r) {
        float inv = 1.0f / l[r];
        int row = q0 + r + 8 * half;
        size_t base = ((size_t)b * SEQ + row) * DIM + (size_t)h * HDIM;
        #pragma unroll
        for (int c = 0; c < 4; ++c)
            Ob[base + c * 16 + n] = f2bf(Oacc[c][r] * inv);
    }
}

// ------------------------------------------------------------- proj GEMM
__global__ void proj_gemm(const unsigned short* __restrict__ ab,
                          const unsigned short* __restrict__ wb,
                          const float* __restrict__ bias,
                          float* __restrict__ out) {
    __shared__ __align__(16) unsigned short ldsA[2 * 2 * 16 * 32];  // 4 KB
    int wave = threadIdx.x >> 5, lane = threadIdx.x & 31;
    int row0 = blockIdx.x * 16;
    int colg = __builtin_amdgcn_readfirstlane(blockIdx.y * 4 + wave); // [0,12)
    int col0 = colg * 64;

    f32x8 acc[4] = {};
    gemm_tile(ab, wb, ldsA, row0, col0, lane, acc);

    int n = lane & 15, half = lane >> 4;
    #pragma unroll
    for (int t = 0; t < 4; ++t) {
        int col = col0 + t * 16 + n;
        float bv = bias[col];
        #pragma unroll
        for (int r = 0; r < 8; ++r) {
            int row = row0 + r + 8 * half;
            out[(size_t)row * DIM + col] = acc[t][r] + bv;
        }
    }
}

// ---------------------------------------------------------------- launcher
extern "C" void kernel_launch(void* const* d_in, const int* in_sizes, int n_in,
                              void* d_out, int out_size, void* d_ws, size_t ws_size,
                              hipStream_t stream) {
    (void)in_sizes; (void)n_in; (void)out_size; (void)ws_size;
    const float* x      = (const float*)d_in[0];
    const float* w_qkv  = (const float*)d_in[1];
    const float* w_proj = (const float*)d_in[2];
    const float* b_proj = (const float*)d_in[3];
    float* out = (float*)d_out;

    char* ws = (char*)d_ws;
    size_t off = 0;
    auto carve = [&](size_t bytes) -> char* {
        char* p = ws + off;
        off += (bytes + 255) & ~(size_t)255;
        return p;
    };

    unsigned short* xb   = (unsigned short*)carve((size_t)BN * DIM * 2);
    unsigned short* wqb  = (unsigned short*)carve((size_t)QKV_O * DIM * 2);
    unsigned short* wpb  = (unsigned short*)carve((size_t)DIM * DIM * 2);
    unsigned short* Qb   = (unsigned short*)carve((size_t)BATCH * NHEADS * SEQ * HDIM * 2);
    unsigned short* Kb   = (unsigned short*)carve((size_t)BATCH * NHEADS * SEQ * HDIM * 2);
    unsigned short* Vb   = (unsigned short*)carve((size_t)BATCH * NHEADS * SEQ * HDIM * 2);
    unsigned short* attn = (unsigned short*)carve((size_t)BN * DIM * 2);

    cast_f32_bf16<<<dim3(1024), 256, 0, stream>>>(x, xb, BN * DIM);
    cast_f32_bf16<<<dim3(512),  256, 0, stream>>>(w_qkv, wqb, QKV_O * DIM);
    cast_f32_bf16<<<dim3(256),  256, 0, stream>>>(w_proj, wpb, DIM * DIM);

    qkv_gemm<<<dim3(BN / 16, 9), 128, 0, stream>>>(xb, wqb, Qb, Kb, Vb);
    flash_attn<<<dim3(SEQ / 64, NHEADS, BATCH), 128, 0, stream>>>(Qb, Kb, Vb, attn);
    proj_gemm<<<dim3(BN / 16, 3), 128, 0, stream>>>(attn, wpb, b_proj, out);
}
